// GPDT_V2_Torch_28192165331059
// MI455X (gfx1250) — matrix operand order repaired and checked by reading, not compiled, not run on hardware
//
#include <hip/hip_runtime.h>
#include <hip/hip_bf16.h>
#include <math.h>

typedef __bf16 bf16;
typedef __attribute__((ext_vector_type(16))) __bf16 v16bf;
typedef __attribute__((ext_vector_type(8)))  float  v8f;

union AFrag { v16bf v; bf16 e[16]; uint4 q[2]; };
union CFrag { v8f  v; float e[8]; };

#define NUM_NODE   9
#define MAX_LEN    60
#define PROMPT_LEN 20
#define TOTAL_T    80
#define NTOK       720
#define HID        256
#define NLAYER     4
#define BATCHSZ    64
#define FFDIM      1024

// ---------------- WMMA + fragment helpers (CDNA5 bf16 16x16x32) ----------------

__device__ inline v8f wmma_bf16(v16bf a, v16bf b, v8f c) {
  return __builtin_amdgcn_wmma_f32_16x16x32_bf16(false, a, false, b, (short)0, c,
                                                 false, false);
}

// A-fragment 16x32 bf16 from row-major [*, lda] source, A points at row0.
// Per lane this is 16 contiguous bf16 (two 16B chunks):
//   K = kg..kg+15, kg = (lane>=16)*16.
__device__ inline v16bf load_a_frag(const bf16* __restrict__ A, int lda, int k0) {
  int lane = threadIdx.x & 31;
  const bf16* p = A + (size_t)(lane & 15) * lda + k0 + ((lane >> 4) * 16);
  AFrag f;
  f.q[0] = *reinterpret_cast<const uint4*>(p);
  f.q[1] = *reinterpret_cast<const uint4*>(p + 8);
  return f.v;
}

// B-fragment 32x16 where logical B[k][n] = M[n][k] with M row-major [*, ldm]
// (transposed weights, K-matrix, transposed V). Per lane this is 16
// *contiguous* bf16: k = k0 + kg .. k0 + kg + 15, kg = (lane>=16)*16.
__device__ inline v16bf load_bt_frag(const bf16* __restrict__ M, int ldm, int n0, int k0) {
  int lane = threadIdx.x & 31;
  const bf16* p = M + (size_t)(n0 + (lane & 15)) * ldm + k0 + ((lane >> 4) * 16);
  AFrag f;
  f.q[0] = *reinterpret_cast<const uint4*>(p);
  f.q[1] = *reinterpret_cast<const uint4*>(p + 8);
  return f.v;
}

__device__ inline float wave_sum(float x) {
#pragma unroll
  for (int off = 16; off > 0; off >>= 1) x += __shfl_xor(x, off, 32);
  return x;
}

// ---------------- weight conversion (f32 [K,N] -> bf16 [N,K]) ----------------

__global__ void k_cvtT(const float* __restrict__ s, bf16* __restrict__ d,
                       int K, int N) {
  int i = blockIdx.x * blockDim.x + threadIdx.x;
  if (i >= K * N) return;
  int k = i / N, n = i % N;
  d[(size_t)n * K + k] = (bf16)s[i];
}

__global__ void k_num_zeros(const int* __restrict__ am, const int* __restrict__ pam,
                            int* __restrict__ nz) {
  int b = threadIdx.x;
  if (b >= BATCHSZ) return;
  int s = 0;
  for (int i = 0; i < MAX_LEN; ++i)    s += am[b * MAX_LEN + i];
  for (int i = 0; i < PROMPT_LEN; ++i) s += pam[b * PROMPT_LEN + i];
  int z = TOTAL_T - s;
  if (z < 0) z = 0;
  if (z > MAX_LEN - 1) z = MAX_LEN - 1;
  nz[b] = z;
}

// ---------------- embedding + LN (one wave per token, shuffle reductions) ------

__global__ void k_embed(
    const float* __restrict__ states,  const float* __restrict__ actions,
    const float* __restrict__ rtg,     const int* __restrict__ tsteps,
    const float* __restrict__ pstates, const float* __restrict__ pactions,
    const float* __restrict__ prtg,    const int* __restrict__ ptsteps,
    const float* __restrict__ root_w,  const float* __restrict__ root_b,
    const float* __restrict__ limb_w,  const float* __restrict__ limb_b,
    const float* __restrict__ ret_w,   const float* __restrict__ ret_b,
    const float* __restrict__ ttab,
    const float* __restrict__ proot_w, const float* __restrict__ proot_b,
    const float* __restrict__ plimb_w, const float* __restrict__ plimb_b,
    const float* __restrict__ pret_w,  const float* __restrict__ pret_b,
    const float* __restrict__ pttab,
    const float* __restrict__ ln_g,    const float* __restrict__ ln_b,
    float* __restrict__ X, bf16* __restrict__ XB) {
  int tok  = blockIdx.x % NTOK;
  int b    = blockIdx.x / NTOK;
  int s    = tok / NUM_NODE;
  int node = tok % NUM_NODE;
  int lane = threadIdx.x & 31;
  bool pr  = s < PROMPT_LEN;
  int sl   = pr ? s : s - PROMPT_LEN;

  const float* st = pr ? (pstates + ((size_t)b * PROMPT_LEN + sl) * 27)
                       : (states  + ((size_t)b * MAX_LEN   + sl) * 27);
  float rv = pr ? prtg[(size_t)b * PROMPT_LEN + sl] : rtg[(size_t)b * MAX_LEN + sl];
  int  ts  = pr ? ptsteps[(size_t)b * PROMPT_LEN + sl] : tsteps[(size_t)b * MAX_LEN + sl];
  const float* rb = pr ? proot_b : root_b;
  const float* lb = pr ? plimb_b : limb_b;
  const float* qw = pr ? pret_w  : ret_w;
  const float* qb = pr ? pret_b  : ret_b;
  const float* tt = pr ? pttab   : ttab;
  const float* W  = (node == 0) ? (pr ? proot_w : root_w) : (pr ? plimb_w : limb_w);
  const float* Bv = (node == 0) ? rb : lb;

  float in0, in1, in2, in3;
  if (node == 0) {
    in0 = st[0]; in1 = st[1]; in2 = st[2]; in3 = 0.0f;
  } else {
    in0 = st[node * 3 + 0]; in1 = st[node * 3 + 1]; in2 = st[node * 3 + 2];
    in3 = pr ? pactions[((size_t)b * (PROMPT_LEN + 1) + sl) * 8 + node - 1]
             : actions [((size_t)b * (MAX_LEN   + 1) + sl) * 8 + node - 1];
  }

  float e[8];
  float psum = 0.0f;
#pragma unroll
  for (int j = 0; j < 8; ++j) {
    int h = lane + 32 * j;
    float v = Bv[h] + in0 * W[h] + in1 * W[HID + h] + in2 * W[2 * HID + h];
    if (node != 0) v += in3 * W[3 * HID + h];
    v += rv * qw[h] + qb[h] + tt[(size_t)ts * HID + h];
    e[j] = v;
    psum += v;
  }
  float mean = wave_sum(psum) * (1.0f / HID);
  float pvar = 0.0f;
#pragma unroll
  for (int j = 0; j < 8; ++j) { float d = e[j] - mean; pvar += d * d; }
  float rstd = rsqrtf(wave_sum(pvar) * (1.0f / HID) + 1e-5f);

  size_t base = ((size_t)b * NTOK + tok) * HID;
#pragma unroll
  for (int j = 0; j < 8; ++j) {
    int h = lane + 32 * j;
    float xv = (e[j] - mean) * rstd * ln_g[h] + ln_b[h];
    X[base + h]  = xv;
    XB[base + h] = (bf16)xv;
  }
}

__global__ void k_ln(const float* __restrict__ Y, const float* __restrict__ g,
                     const float* __restrict__ bb, float* __restrict__ X,
                     bf16* __restrict__ XB) {
  int lane = threadIdx.x & 31;
  size_t base = (size_t)blockIdx.x * HID;
  float e[8];
  float psum = 0.0f;
#pragma unroll
  for (int j = 0; j < 8; ++j) {
    e[j] = Y[base + lane + 32 * j];
    psum += e[j];
  }
  float mean = wave_sum(psum) * (1.0f / HID);
  float pvar = 0.0f;
#pragma unroll
  for (int j = 0; j < 8; ++j) { float d = e[j] - mean; pvar += d * d; }
  float rstd = rsqrtf(wave_sum(pvar) * (1.0f / HID) + 1e-5f);
#pragma unroll
  for (int j = 0; j < 8; ++j) {
    int h = lane + 32 * j;
    float xv = (e[j] - mean) * rstd * g[h] + bb[h];
    X[base + h]  = xv;
    XB[base + h] = (bf16)xv;
  }
}

// ---------------- WMMA GEMM, 256x256 weights, transposed WT[N,K] --------------
// grid.x = M/16, block = 128 (4 waves); wave w covers columns w*64..w*64+63.
// MODE 0: out bf16 row-major. MODE 1: out bf16 transposed per batch
// ([B,HID,NTOK], for V). MODE 2: out f32 + residual.
template <int MODE>
__global__ void k_gemm(const bf16* __restrict__ A, const bf16* __restrict__ WT,
                       const float* __restrict__ bias,
                       const float* __restrict__ resid,
                       float* __restrict__ outF, bf16* __restrict__ outB) {
  const int K = HID, N = HID;
  int wave = threadIdx.x >> 5;
  int lane = threadIdx.x & 31;
  int row0 = blockIdx.x * 16;
  int n0w  = wave * 64;

  CFrag acc[4];
#pragma unroll
  for (int t = 0; t < 4; ++t)
#pragma unroll
    for (int j = 0; j < 8; ++j) acc[t].e[j] = 0.0f;

  const bf16* Ar = A + (size_t)row0 * K;
#pragma unroll
  for (int k0 = 0; k0 < K; k0 += 32) {
    if (k0 + 32 < K)
      __builtin_prefetch(&WT[(size_t)n0w * K + k0 + 32], 0, 2);
    v16bf a = load_a_frag(Ar, K, k0);
#pragma unroll
    for (int t = 0; t < 4; ++t) {
      v16bf bf = load_bt_frag(WT, K, n0w + 16 * t, k0);
      acc[t].v = wmma_bf16(a, bf, acc[t].v);
    }
  }

  int hi = lane >> 4, nn = lane & 15;
  int bidx = row0 / NTOK;
  int tok0 = row0 % NTOK;
#pragma unroll
  for (int t = 0; t < 4; ++t) {
    int n = n0w + 16 * t + nn;
    float bv = bias[n];
#pragma unroll
    for (int r = 0; r < 8; ++r) {
      int m = r + 8 * hi;
      size_t row = (size_t)(row0 + m);
      float v = acc[t].e[r] + bv;
      if (MODE == 2) {
        v += resid[row * N + n];
        outF[row * N + n] = v;
      } else if (MODE == 0) {
        outB[row * N + n] = (bf16)v;
      } else {
        outB[((size_t)bidx * HID + n) * NTOK + tok0 + m] = (bf16)v;
      }
    }
  }
}

// ---------------- flash-style attention (1 head, d=256) ----------------
// grid = (45 q-tiles, BATCH), block = 32. V is transposed [B,HID,NTOK].
// launch_bounds(32,2): ~512 VGPR budget so aq[8] + o[16] stay in registers
// (round-2 build spilled the Q fragments to scratch inside the hot loop).
__global__ void __launch_bounds__(32, 2)
k_attn(const bf16* __restrict__ Q, const bf16* __restrict__ Kb,
       const bf16* __restrict__ VT,
       const unsigned char* __restrict__ mstore,
       const int* __restrict__ nz, bf16* __restrict__ CTX) {
  int b    = blockIdx.y;
  int row0 = blockIdx.x * 16;
  int lane = threadIdx.x & 31;
  int hi   = lane >> 4, nn = lane & 15;

  const bf16* Qr    = Q  + ((size_t)b * NTOK + row0) * HID;
  const bf16* Kbase = Kb + (size_t)b * NTOK * HID;
  const bf16* Vbase = VT + (size_t)b * HID * NTOK;
  const unsigned char* mrow = mstore + (size_t)nz[b] * NTOK * NTOK;

  v16bf aq[8];
#pragma unroll
  for (int i = 0; i < 8; ++i) aq[i] = load_a_frag(Qr, HID, i * 32);

  float mrun[8], lrun[8];
#pragma unroll
  for (int r = 0; r < 8; ++r) { mrun[r] = -1e30f; lrun[r] = 0.0f; }

  CFrag o[16];
#pragma unroll
  for (int t = 0; t < 16; ++t)
#pragma unroll
    for (int j = 0; j < 8; ++j) o[t].e[j] = 0.0f;

  __shared__ __align__(16) bf16 pl[16 * 32];

  for (int kt = 0; kt < 23; ++kt) {
    int key0 = kt * 32;
    bool have1 = (key0 + 16) < NTOK;

    CFrag s0, s1;
#pragma unroll
    for (int j = 0; j < 8; ++j) { s0.e[j] = 0.0f; s1.e[j] = 0.0f; }
#pragma unroll
    for (int kk = 0; kk < 8; ++kk)
      s0.v = wmma_bf16(aq[kk], load_bt_frag(Kbase, HID, key0, kk * 32), s0.v);
    if (have1) {
#pragma unroll
      for (int kk = 0; kk < 8; ++kk)
        s1.v = wmma_bf16(aq[kk], load_bt_frag(Kbase, HID, key0 + 16, kk * 32), s1.v);
    }

    float corr[8];
#pragma unroll
    for (int r = 0; r < 8; ++r) {
      int m = r + 8 * hi;
      int qrow = row0 + m;
      float v0 = s0.e[r] * 0.0625f;
      if (mrow[(size_t)qrow * NTOK + key0 + nn]) v0 = -1e30f;
      float v1 = -1e30f;
      if (have1) {
        v1 = s1.e[r] * 0.0625f;
        if (mrow[(size_t)qrow * NTOK + key0 + 16 + nn]) v1 = -1e30f;
      }
      float mx = fmaxf(v0, v1);
#pragma unroll
      for (int off = 1; off < 16; off <<= 1) mx = fmaxf(mx, __shfl_xor(mx, off, 32));
      float nm = fmaxf(mrun[r], mx);
      corr[r] = __expf(mrun[r] - nm);
      float p0 = (v0 > -1e29f) ? __expf(v0 - nm) : 0.0f;
      float p1 = (v1 > -1e29f) ? __expf(v1 - nm) : 0.0f;
      float ts = p0 + p1;
#pragma unroll
      for (int off = 1; off < 16; off <<= 1) ts += __shfl_xor(ts, off, 32);
      lrun[r] = lrun[r] * corr[r] + ts;
      mrun[r] = nm;
      s0.e[r] = p0;
      s1.e[r] = p1;
    }

#pragma unroll
    for (int t = 0; t < 16; ++t)
#pragma unroll
      for (int r = 0; r < 8; ++r) o[t].e[r] *= corr[r];

#pragma unroll
    for (int r = 0; r < 8; ++r) {
      int m = r + 8 * hi;
      pl[m * 32 + nn]      = (bf16)s0.e[r];
      pl[m * 32 + 16 + nn] = (bf16)s1.e[r];
    }
    __syncthreads();
    v16bf pa = load_a_frag(pl, 32, 0);
    __syncthreads();

    // P @ V: B[k=key][n=chan] = VT[chan][key]; per lane 16 contiguous keys.
    int kg = hi * 16;
    bool vok = (key0 + kg) < NTOK;
#pragma unroll
    for (int ct = 0; ct < 16; ++ct) {
      AFrag fb;
      if (vok) {
        const bf16* p = Vbase + (size_t)(ct * 16 + nn) * NTOK + key0 + kg;
        fb.q[0] = *reinterpret_cast<const uint4*>(p);
        fb.q[1] = *reinterpret_cast<const uint4*>(p + 8);
      } else {
        fb.q[0] = make_uint4(0, 0, 0, 0);
        fb.q[1] = make_uint4(0, 0, 0, 0);
      }
      o[ct].v = wmma_bf16(pa, fb.v, o[ct].v);
    }
  }

  bf16* cb = CTX + ((size_t)b * NTOK + row0) * HID;
#pragma unroll
  for (int r = 0; r < 8; ++r) {
    float inv = (lrun[r] > 0.0f) ? 1.0f / lrun[r] : 0.0f;
    int m = r + 8 * hi;
#pragma unroll
    for (int ct = 0; ct < 16; ++ct)
      cb[(size_t)m * HID + ct * 16 + nn] = (bf16)(o[ct].e[r] * inv);
  }
}

// ---------------- fused FFN: relu(x@W1+b1)@W2 + b2 + resid ----------------
// grid = (45, BATCH), block = 32. W1T [1024,256], W2T [256,1024].
__global__ void __launch_bounds__(32, 2)
k_ffn(const bf16* __restrict__ XBin, const bf16* __restrict__ W1T,
      const float* __restrict__ b1, const bf16* __restrict__ W2T,
      const float* __restrict__ b2, const float* __restrict__ Xres,
      float* __restrict__ Y) {
  int b    = blockIdx.y;
  int row0 = blockIdx.x * 16;
  size_t R = (size_t)b * NTOK + row0;
  int lane = threadIdx.x & 31;
  int hi   = lane >> 4, nn = lane & 15;

  v16bf ax[8];
#pragma unroll
  for (int i = 0; i < 8; ++i) ax[i] = load_a_frag(XBin + R * HID, HID, i * 32);

  CFrag o[16];
#pragma unroll
  for (int t = 0; t < 16; ++t)
#pragma unroll
    for (int j = 0; j < 8; ++j) o[t].e[j] = 0.0f;

  __shared__ __align__(16) bf16 hl[16 * 32];

  for (int j2 = 0; j2 < FFDIM / 32; ++j2) {
    int h0 = j2 * 32;
    CFrag h0t, h1t;
#pragma unroll
    for (int j = 0; j < 8; ++j) { h0t.e[j] = 0.0f; h1t.e[j] = 0.0f; }
#pragma unroll
    for (int kk = 0; kk < 8; ++kk) {
      h0t.v = wmma_bf16(ax[kk], load_bt_frag(W1T, HID, h0,      kk * 32), h0t.v);
      h1t.v = wmma_bf16(ax[kk], load_bt_frag(W1T, HID, h0 + 16, kk * 32), h1t.v);
    }
#pragma unroll
    for (int r = 0; r < 8; ++r) {
      int m = r + 8 * hi;
      hl[m * 32 + nn]      = (bf16)fmaxf(h0t.e[r] + b1[h0 + nn], 0.0f);
      hl[m * 32 + 16 + nn] = (bf16)fmaxf(h1t.e[r] + b1[h0 + 16 + nn], 0.0f);
    }
    __syncthreads();
    v16bf pa = load_a_frag(hl, 32, 0);
    __syncthreads();
#pragma unroll
    for (int ct = 0; ct < 16; ++ct)
      o[ct].v = wmma_bf16(pa, load_bt_frag(W2T, FFDIM, ct * 16, h0), o[ct].v);
  }

#pragma unroll
  for (int r = 0; r < 8; ++r) {
    int m = r + 8 * hi;
    size_t row = R + m;
#pragma unroll
    for (int ct = 0; ct < 16; ++ct) {
      int n = ct * 16 + nn;
      Y[row * HID + n] = o[ct].e[r] + b2[n] + Xres[row * HID + n];
    }
  }
}

// ---------------- prediction head ----------------

__global__ void k_head(const float* __restrict__ X, const float* __restrict__ w,
                       const float* __restrict__ bsc, float* __restrict__ out) {
  int b = blockIdx.x / MAX_LEN;
  int s = blockIdx.x % MAX_LEN;
  int limb = threadIdx.x >> 5;
  int lane = threadIdx.x & 31;
  int token = (PROMPT_LEN + s) * NUM_NODE + limb + 1;
  const float* xr = X + ((size_t)b * NTOK + token) * HID;
  float acc = 0.0f;
#pragma unroll
  for (int j = 0; j < 8; ++j) acc += xr[lane + 32 * j] * w[lane + 32 * j];
  acc = wave_sum(acc);
  if (lane == 0) out[((size_t)b * MAX_LEN + s) * 8 + limb] = tanhf(acc + bsc[0]);
}

// ---------------- host driver ----------------

extern "C" void kernel_launch(void* const* d_in, const int* in_sizes, int n_in,
                              void* d_out, int out_size, void* d_ws, size_t ws_size,
                              hipStream_t stream) {
  (void)in_sizes; (void)n_in; (void)out_size; (void)ws_size;
#define INF(i) ((const float*)d_in[(i)])
#define INI(i) ((const int*)d_in[(i)])
  const float* states   = INF(0);
  const float* actions  = INF(1);
  const float* rtg      = INF(2);
  const int*   tsteps   = INI(3);
  const int*   amask    = INI(4);
  const float* pstates  = INF(5);
  const float* pactions = INF(6);
  const float* prtg     = INF(7);
  const int*   ptsteps  = INI(8);
  const int*   pamask   = INI(9);
  const unsigned char* mask_store = (const unsigned char*)d_in[10];

  const float* limb_b = INF(11); const float* limb_w = INF(12);
  const float* ln_b   = INF(13); const float* ln_g   = INF(14);
  const float* ret_b  = INF(15); const float* ret_w  = INF(16);
  const float* root_b = INF(17); const float* root_w = INF(18);
  const float* ttab   = INF(19);
  const float* plimb_b = INF(84); const float* plimb_w = INF(85);
  const float* pret_b  = INF(86); const float* pret_w  = INF(87);
  const float* proot_b = INF(88); const float* proot_w = INF(89);
  const float* pttab   = INF(90);
  const float* pred_b  = INF(91); const float* pred_w  = INF(92);

  const size_t LSTRIDE = 4 * 65536 + 2 * 262144;  // bf16 elems per layer
  char* ws = (char*)d_ws;
  bf16* WB = (bf16*)ws;
  size_t off = NLAYER * LSTRIDE * 2;
  off = (off + 255) & ~(size_t)255;
  const size_t ACT_F = (size_t)BATCHSZ * NTOK * HID * 4;
  const size_t ACT_B = (size_t)BATCHSZ * NTOK * HID * 2;
  float* X  = (float*)(ws + off); off += ACT_F;
  bf16*  XB = (bf16*) (ws + off); off += ACT_B;
  bf16*  Qb = (bf16*) (ws + off); off += ACT_B;
  bf16*  Kb = (bf16*) (ws + off); off += ACT_B;
  bf16*  Vt = (bf16*) (ws + off); off += ACT_B;  // V stored transposed [B,HID,NTOK]
  float* Y  = (float*)(ws + off); off += ACT_F;
  int*   NZ = (int*)  (ws + off);

  // convert + transpose layer weights to bf16 [N,K]
  for (int L = 0; L < NLAYER; ++L) {
    int base = 20 + 16 * L;
    bf16* WL = WB + (size_t)L * LSTRIDE;
    struct { const float* s; bf16* d; int K, N; } cv[6] = {
      { INF(base + 14), WL + 0,      HID,   HID   },  // wq
      { INF(base + 12), WL + 65536,  HID,   HID   },  // wk
      { INF(base + 15), WL + 131072, HID,   HID   },  // wv
      { INF(base + 13), WL + 196608, HID,   HID   },  // wo
      { INF(base + 10), WL + 262144, HID,   FFDIM },  // w1 -> [1024,256]
      { INF(base + 11), WL + 524288, FFDIM, HID   },  // w2 -> [256,1024]
    };
    for (int i = 0; i < 6; ++i) {
      int n = cv[i].K * cv[i].N;
      k_cvtT<<<dim3((n + 255) / 256), dim3(256), 0, stream>>>(cv[i].s, cv[i].d,
                                                              cv[i].K, cv[i].N);
    }
  }

  k_num_zeros<<<dim3(1), dim3(64), 0, stream>>>(amask, pamask, NZ);

  const int NROW = BATCHSZ * NTOK;  // 46080
  k_embed<<<dim3(NROW), dim3(32), 0, stream>>>(
      states, actions, rtg, tsteps, pstates, pactions, prtg, ptsteps,
      root_w, root_b, limb_w, limb_b, ret_w, ret_b, ttab,
      proot_w, proot_b, plimb_w, plimb_b, pret_w, pret_b, pttab,
      ln_g, ln_b, X, XB);

  for (int L = 0; L < NLAYER; ++L) {
    int base = 20 + 16 * L;
    bf16* WL = WB + (size_t)L * LSTRIDE;
    const float* b1    = INF(base + 0);
    const float* b2    = INF(base + 1);
    const float* bk    = INF(base + 2);
    const float* bo    = INF(base + 3);
    const float* bq    = INF(base + 4);
    const float* bv    = INF(base + 5);
    const float* ln1_b = INF(base + 6);
    const float* ln1_g = INF(base + 7);
    const float* ln2_b = INF(base + 8);
    const float* ln2_g = INF(base + 9);

    dim3 ggrid(NROW / 16);
    k_gemm<0><<<ggrid, dim3(128), 0, stream>>>(XB, WL + 0,      bq, nullptr,
                                               nullptr, Qb);
    k_gemm<0><<<ggrid, dim3(128), 0, stream>>>(XB, WL + 65536,  bk, nullptr,
                                               nullptr, Kb);
    k_gemm<1><<<ggrid, dim3(128), 0, stream>>>(XB, WL + 131072, bv, nullptr,
                                               nullptr, Vt);

    k_attn<<<dim3(NTOK / 16, BATCHSZ), dim3(32), 0, stream>>>(
        Qb, Kb, Vt, mask_store, NZ, Qb);

    k_gemm<2><<<ggrid, dim3(128), 0, stream>>>(Qb, WL + 196608, bo, X, Y, nullptr);
    k_ln<<<dim3(NROW), dim3(32), 0, stream>>>(Y, ln1_g, ln1_b, X, XB);

    k_ffn<<<dim3(NTOK / 16, BATCHSZ), dim3(32), 0, stream>>>(
        XB, WL + 262144, b1, WL + 524288, b2, X, Y);
    k_ln<<<dim3(NROW), dim3(32), 0, stream>>>(Y, ln2_g, ln2_b, X, XB);
  }

  k_head<<<dim3(BATCHSZ * MAX_LEN), dim3(256), 0, stream>>>(
      X, pred_w, pred_b, (float*)d_out);
#undef INF
#undef INI
}
